// DeformConv3d_12317966204943
// MI455X (gfx1250) — matrix operand order repaired and checked by reading, not compiled, not run on hardware
//
#include <hip/hip_runtime.h>

#define H 64
#define W 64
#define HW 4096
#define NIMG 32
#define KK 270
#define KPAD 288
#define NT 64
#define SPEC_K 48
#define SPEC_KPAD 64
#define SPEC_NT 128
#define NBN (2*30*HW)   // 245760 elements per BN channel

typedef __attribute__((ext_vector_type(16))) __bf16 v16bf;
typedef __attribute__((ext_vector_type(8)))  float  v8f;

union Frag16 { v16bf v; uint4 q[2]; unsigned short s[16]; };

__device__ __forceinline__ unsigned short f2bf(float f) {
    unsigned int u = __float_as_uint(f);
    u += 0x7fffu + ((u >> 16) & 1u);          // round-to-nearest-even
    return (unsigned short)(u >> 16);
}

// ---------------- weight packing / folding (runs once per launch) ----------
__global__ void k_pack(const float* __restrict__ p_w, const float* __restrict__ m_w,
                       const float* __restrict__ conv_w, const float* __restrict__ comp_w,
                       const float* __restrict__ spec_w,
                       unsigned short* __restrict__ A_pm, unsigned short* __restrict__ A2,
                       unsigned short* __restrict__ A_spec, float* __restrict__ stats) {
    int tid = threadIdx.x;
    if (tid < 32) stats[tid] = 0.f;
    for (int idx = tid; idx < 32 * KPAD; idx += blockDim.x) {
        int m = idx / KPAD, k = idx % KPAD;
        float v = 0.f;
        if (k < KK) {
            if (m < 18)      v = p_w[m * KK + k];
            else if (m < 27) v = m_w[(m - 18) * KK + k];
        }
        A_pm[idx] = f2bf(v);
    }
    // Fold comp (30x90 1x1) into conv (90x270): W2 = comp * conv  (both linear)
    for (int idx = tid; idx < 32 * KPAD; idx += blockDim.x) {
        int m = idx / KPAD, k = idx % KPAD;
        float v = 0.f;
        if (m < 30 && k < KK) {
            float acc = 0.f;
            for (int o = 0; o < 90; ++o) acc += comp_w[m * 90 + o] * conv_w[o * KK + k];
            v = acc;
        }
        A2[idx] = f2bf(v);
    }
    for (int idx = tid; idx < 16 * SPEC_KPAD; idx += blockDim.x) {
        int m = idx / SPEC_KPAD, k = idx % SPEC_KPAD;
        A_spec[idx] = f2bf(k < SPEC_K ? spec_w[m * SPEC_K + k] : 0.f);
    }
}

// ------- replicate the reference's transpose(0,2,1,3,4).reshape(32,30,...) --
__global__ void k_xr(const float* __restrict__ x, float* __restrict__ xr) {
    int i = (blockIdx.x * blockDim.x + threadIdx.x) * 4;   // 32*30*4096 total
    int p = i & (HW - 1);
    int f = i >> 12;                 // bt*30 + c over flattened (B,C,T)
    int b = f / 480;
    int rem = f % 480;
    int c_orig = rem / 16;
    int t = rem % 16;
    *(float4*)(xr + i) = *(const float4*)(x + ((b * 16 + t) * 30 + c_orig) * HW + p);
}

// -------------------- shared WMMA 16x16 tile GEMM core ---------------------
__device__ __forceinline__ v8f gemm_tile288(const unsigned short* __restrict__ A_sm,
                                            const unsigned short* __restrict__ col_sm,
                                            int mt, int nt, int lane) {
    v8f acc = {};
    int m16 = lane & 15, half = lane >> 4;
    const unsigned short* Arow = A_sm  + (mt * 16 + m16) * KPAD;
    const unsigned short* Brow = col_sm + (nt * 16 + m16) * KPAD;
    for (int kb = 0; kb < KPAD; kb += 32) {
        Frag16 a, b;
        a.q[0] = *(const uint4*)(Arow + kb + 8 * half);        // A: e0..7 -> K=kb+8h..+7
        a.q[1] = *(const uint4*)(Arow + kb + 16 + 8 * half);   // A: e8..15 -> K=kb+16+8h..
        b.q[0] = *(const uint4*)(Brow + kb + 8 * half);        // B: e0..7 -> K=kb+8h..+7
        b.q[1] = *(const uint4*)(Brow + kb + 16 + 8 * half);   // B: e8..15 -> K=kb+16+8h..
        acc = __builtin_amdgcn_wmma_f32_16x16x32_bf16(false, a.v, false, b.v,
                                                      (short)0, acc, false, false);
    }
    return acc;
}

// --- fused: offset/mask conv (27x270 GEMM, sigmoid) kept in LDS, then
// --- deformable bilinear im2col + folded 30x270 GEMM -> out30 --------------
__global__ __launch_bounds__(256) void k_fused(const float* __restrict__ xr,
                                               const unsigned short* __restrict__ A_pm,
                                               const unsigned short* __restrict__ A2,
                                               const float* __restrict__ p_b,
                                               const float* __restrict__ m_b,
                                               float* __restrict__ out30) {
    __shared__ __align__(16) unsigned short As[32 * KPAD];    // 18 KB (reused)
    __shared__ __align__(16) unsigned short col[NT * KPAD];   // 36 KB (reused)
    __shared__ float omt[27 * NT];                            // 6.75 KB offsets+masks
    int bt = blockIdx.x >> 6;
    int p0 = (blockIdx.x & 63) * NT;
    int tid = threadIdx.x;
    const float* img = xr + bt * 30 * HW;

    // ---- stage 1: plain im2col of x, GEMM with packed p_w/m_w -------------
    for (int i = tid; i < 32 * KPAD; i += 256) As[i] = A_pm[i];
    for (int idx = tid; idx < KPAD * NT; idx += 256) {
        int k = idx / NT, n = idx % NT;
        float v = 0.f;
        if (k < KK) {
            int c = k / 9, t = k % 9;
            int p = p0 + n;
            int y = (p >> 6) + t / 3 - 1, x = (p & 63) + t % 3 - 1;
            if ((unsigned)y < (unsigned)H && (unsigned)x < (unsigned)W)
                v = img[c * HW + y * W + x];
        }
        col[n * KPAD + k] = f2bf(v);
    }
    __syncthreads();
    int wid = tid >> 5, lane = tid & 31;
    int mt = wid >> 2, nt = wid & 3;
    int m16 = lane & 15, half = lane >> 4;
    {
        v8f acc = gemm_tile288(As, col, mt, nt, lane);
        int n = nt * 16 + m16;
        for (int r = 0; r < 8; ++r) {
            int ch = mt * 16 + r + 8 * half;
            if (ch < 27) {
                float v = acc[r];
                if (ch < 18) v += p_b[ch];
                else { v += m_b[ch - 18]; v = 1.f / (1.f + __expf(-v)); }
                omt[ch * NT + n] = v;
            }
        }
    }
    __syncthreads();   // everyone done reading As/col of stage 1

    // ---- stage 2: deform im2col from omt (LDS), GEMM with folded W2 -------
    for (int i = tid; i < 32 * KPAD; i += 256) As[i] = A2[i];
    for (int i = tid; i < NT * (KPAD - KK); i += 256) {
        int n = i / (KPAD - KK), k = KK + i % (KPAD - KK);
        col[n * KPAD + k] = 0;
    }
    for (int idx = tid; idx < 9 * NT; idx += 256) {
        int t = idx / NT, n = idx % NT;
        int p = p0 + n;
        int y = p >> 6, x = p & 63;
        float offx = omt[t * NT + n];
        float offy = omt[(9 + t) * NT + n];
        float msk  = omt[(18 + t) * NT + n];
        float px = offx + (float)(t / 3 - 1) + (float)(y + 1);
        float py = offy + (float)(t % 3 - 1) + (float)(x + 1);
        float fx = floorf(px), fy = floorf(py);
        int ltx = min(max((int)fx, 0), 65),     lty = min(max((int)fy, 0), 65);
        int rbx = min(max((int)fx + 1, 0), 65), rby = min(max((int)fy + 1, 0), 65);
        float pxc = fminf(fmaxf(px, 0.f), 65.f);
        float pyc = fminf(fmaxf(py, 0.f), 65.f);
        float glt = (1.f + ((float)ltx - pxc)) * (1.f + ((float)lty - pyc));
        float grb = (1.f - ((float)rbx - pxc)) * (1.f - ((float)rby - pyc));
        float glb = (1.f + ((float)ltx - pxc)) * (1.f - ((float)rby - pyc));
        float grt = (1.f - ((float)rbx - pxc)) * (1.f + ((float)lty - pyc));
        int i_lt = (ltx >= 1 && ltx <= 64 && lty >= 1 && lty <= 64) ? (ltx - 1) * W + (lty - 1) : -1;
        int i_rb = (rbx >= 1 && rbx <= 64 && rby >= 1 && rby <= 64) ? (rbx - 1) * W + (rby - 1) : -1;
        int i_lb = (ltx >= 1 && ltx <= 64 && rby >= 1 && rby <= 64) ? (ltx - 1) * W + (rby - 1) : -1;
        int i_rt = (rbx >= 1 && rbx <= 64 && lty >= 1 && lty <= 64) ? (rbx - 1) * W + (lty - 1) : -1;
        unsigned short* cp = &col[n * KPAD + t];
        for (int c = 0; c < 30; ++c) {
            const float* pl = img + c * HW;
            float s = 0.f;
            if (i_lt >= 0) s += glt * pl[i_lt];
            if (i_rb >= 0) s += grb * pl[i_rb];
            if (i_lb >= 0) s += glb * pl[i_lb];
            if (i_rt >= 0) s += grt * pl[i_rt];
            cp[c * 9] = f2bf(msk * s);
        }
    }
    __syncthreads();
    {
        v8f acc = gemm_tile288(As, col, mt, nt, lane);
        int n = p0 + nt * 16 + m16;
        for (int r = 0; r < 8; ++r) {
            int ch = mt * 16 + r + 8 * half;
            if (ch < 30) out30[(bt * 30 + ch) * HW + n] = acc[r];
        }
    }
}

// ------------- spec conv (16x48 GEMM over T/C) + BN partial sums -----------
__global__ __launch_bounds__(256) void k_spec(const float* __restrict__ out30,
                                              const unsigned short* __restrict__ A_spec,
                                              float* __restrict__ spec,
                                              float* __restrict__ stats) {
    __shared__ __align__(16) unsigned short As[16 * SPEC_KPAD];
    __shared__ __align__(16) unsigned short col[SPEC_NT * SPEC_KPAD];
    __shared__ float ssum[16], ssq[16];
    int tid = threadIdx.x;
    if (tid < 16) { ssum[tid] = 0.f; ssq[tid] = 0.f; }
    for (int i = tid; i < 16 * SPEC_KPAD; i += 256) As[i] = A_spec[i];
    int n0 = blockIdx.x * SPEC_NT;           // flattened (b, c, p)
    int b = n0 / (30 * HW);
    int rem = n0 % (30 * HW);
    int c = rem / HW;
    int pbase = rem % HW;                    // 4096 % 128 == 0 -> one (b,c) per block
    for (int idx = tid; idx < SPEC_NT * SPEC_KPAD; idx += 256) {
        int n = idx % SPEC_NT, k = idx / SPEC_NT;
        float v = 0.f;
        if (k < SPEC_K) {
            int ti = k / 3, cc = c + (k % 3) - 1;
            if (cc >= 0 && cc < 30) v = out30[((b * 16 + ti) * 30 + cc) * HW + pbase + n];
        }
        col[n * SPEC_KPAD + k] = f2bf(v);
    }
    __syncthreads();
    int wid = tid >> 5, lane = tid & 31;
    int m16 = lane & 15, half = lane >> 4;
    v8f acc = {};
    const unsigned short* Arow = As + m16 * SPEC_KPAD;
    const unsigned short* Brow = col + (wid * 16 + m16) * SPEC_KPAD;
    for (int kb = 0; kb < SPEC_KPAD; kb += 32) {
        Frag16 a, bf;
        a.q[0]  = *(const uint4*)(Arow + kb + 8 * half);
        a.q[1]  = *(const uint4*)(Arow + kb + 16 + 8 * half);
        bf.q[0] = *(const uint4*)(Brow + kb + 8 * half);
        bf.q[1] = *(const uint4*)(Brow + kb + 16 + 8 * half);
        acc = __builtin_amdgcn_wmma_f32_16x16x32_bf16(false, a.v, false, bf.v,
                                                      (short)0, acc, false, false);
    }
    int n = n0 + wid * 16 + m16;
    for (int r = 0; r < 8; ++r) {
        int to = r + 8 * half;
        float v = acc[r];
        spec[to * NBN + n] = v;
        // wave32 butterfly over the 16-lane N-group (channel is uniform per group)
        float s = v, q = v * v;
        s += __shfl_xor(s, 1); q += __shfl_xor(q, 1);
        s += __shfl_xor(s, 2); q += __shfl_xor(q, 2);
        s += __shfl_xor(s, 4); q += __shfl_xor(q, 4);
        s += __shfl_xor(s, 8); q += __shfl_xor(q, 8);
        if (m16 == 0) { atomicAdd(&ssum[to], s); atomicAdd(&ssq[to], q); }
    }
    __syncthreads();
    if (tid < 16) {
        atomicAdd(&stats[tid], ssum[tid]);
        atomicAdd(&stats[16 + tid], ssq[tid]);
    }
}

// -------------------------- BN + affine + ReLU -----------------------------
__global__ void k_bn(const float* __restrict__ spec, const float* __restrict__ stats,
                     const float* __restrict__ bn_g, const float* __restrict__ bn_b,
                     float* __restrict__ out) {
    int i = (blockIdx.x * blockDim.x + threadIdx.x) * 4;   // 2*16*30*4096
    int p = i & (HW - 1);
    int f = i >> 12;                 // (b*16+to)*30 + c
    int c = f % 30;
    int bt = f / 30;
    int to = bt & 15;
    int b = bt >> 4;
    int n = (b * 30 + c) * HW + p;
    float inv = 1.f / (float)NBN;
    float mean = stats[to] * inv;
    float var  = stats[16 + to] * inv - mean * mean;
    float rs   = rsqrtf(var + 1e-5f) * bn_g[to];
    float base = bn_b[to] - mean * rs;
    float4 v = *(const float4*)(spec + to * NBN + n);
    float4 o;
    o.x = fmaxf(v.x * rs + base, 0.f);
    o.y = fmaxf(v.y * rs + base, 0.f);
    o.z = fmaxf(v.z * rs + base, 0.f);
    o.w = fmaxf(v.w * rs + base, 0.f);
    *(float4*)(out + i) = o;
}

extern "C" void kernel_launch(void* const* d_in, const int* in_sizes, int n_in,
                              void* d_out, int out_size, void* d_ws, size_t ws_size,
                              hipStream_t stream) {
    const float* x      = (const float*)d_in[0];
    const float* p_w    = (const float*)d_in[1];
    const float* p_b    = (const float*)d_in[2];
    const float* m_w    = (const float*)d_in[3];
    const float* m_b    = (const float*)d_in[4];
    const float* conv_w = (const float*)d_in[5];
    const float* comp_w = (const float*)d_in[6];
    const float* spec_w = (const float*)d_in[7];
    const float* bn_g   = (const float*)d_in[8];
    const float* bn_b   = (const float*)d_in[9];
    float* out = (float*)d_out;

    char* ws = (char*)d_ws;
    size_t off = 0;
    auto alloc = [&](size_t bytes) -> char* {
        char* p = ws + off; off += (bytes + 255) & ~(size_t)255; return p;
    };
    float* xr    = (float*)alloc((size_t)NIMG * 30 * HW * 4);
    float* out30 = (float*)alloc((size_t)NIMG * 30 * HW * 4);
    float* spec  = (float*)alloc((size_t)16 * NBN * 4);
    unsigned short* A_pm   = (unsigned short*)alloc(32 * KPAD * 2);
    unsigned short* A2     = (unsigned short*)alloc(32 * KPAD * 2);
    unsigned short* A_spec = (unsigned short*)alloc(16 * SPEC_KPAD * 2);
    float* stats = (float*)alloc(32 * 4);

    k_pack <<<1,    256, 0, stream>>>(p_w, m_w, conv_w, comp_w, spec_w, A_pm, A2, A_spec, stats);
    k_xr   <<<3840, 256, 0, stream>>>(x, xr);
    k_fused<<<2048, 256, 0, stream>>>(xr, A_pm, A2, p_b, m_b, out30);
    k_spec <<<1920, 256, 0, stream>>>(out30, A_spec, spec, stats);
    k_bn   <<<3840, 256, 0, stream>>>(spec, stats, bn_g, bn_b, out);
}
